// NodeFeatureSelfAttention_34359739014
// MI455X (gfx1250) — matrix operand order repaired and checked by reading, not compiled, not run on hardware
//
#include <hip/hip_runtime.h>
#include <hip/hip_bf16.h>

// ---- CDNA5 vector types ----
typedef _Float16 h4   __attribute__((ext_vector_type(4)));
typedef _Float16 v8h  __attribute__((ext_vector_type(8)));
typedef _Float16 v16h __attribute__((ext_vector_type(16)));
typedef float    v4f  __attribute__((ext_vector_type(4)));
typedef float    v8f  __attribute__((ext_vector_type(8)));

#define DD 128              // feature dim
#define WAVES 2             // waves per block (wave32)
#define RPW 16              // rows (nodes) per wave = WMMA M
#define RPB (WAVES * RPW)   // rows per block

// ------------------------------------------------------------------
// Prep: convert Wq|Wk|Wv (fp32, [128][128] row-major = [out][in]) to f16
// in workspace. Row-major [out][in] is exactly the WMMA B-operand's
// transposed-weight layout for Q = xp @ W^T.
// ------------------------------------------------------------------
__global__ void prep_weights_f16(const float* __restrict__ Wq,
                                 const float* __restrict__ Wk,
                                 const float* __restrict__ Wv,
                                 _Float16* __restrict__ Wh) {
  int i = blockIdx.x * blockDim.x + threadIdx.x;     // 0 .. 3*16384-1
  const float* src = (i < DD * DD) ? Wq : (i < 2 * DD * DD) ? Wk : Wv;
  Wh[i] = (_Float16)src[i % (DD * DD)];
}

// ------------------------------------------------------------------
// Fused: xp = x + pe ; Q,K,V = xp@W^T + b (WMMA f16->f32) ;
// out[n,i] = sum_j softmax_j(q_i * k_j / sqrt(D)) * v_j
// One wave per 16-node tile; QKV staged in LDS.
// ------------------------------------------------------------------
__global__ __launch_bounds__(WAVES * 32)
void node_attn_kernel(const float* __restrict__ x,
                      const _Float16* __restrict__ Wh,  // [3][128][128] f16
                      const float* __restrict__ bq,
                      const float* __restrict__ bk,
                      const float* __restrict__ bv,
                      float* __restrict__ out,
                      int N) {
  __shared__ _Float16 sXp[WAVES][RPW * DD];       // xp tile, f16, row-major
  __shared__ float    sQ [WAVES][RPW * DD];       // Q tile, f32
  __shared__ float    sKV[WAVES][RPW * 2 * DD];   // interleaved (K,V) pairs
  __shared__ float    sPe[DD];                    // positional encoding

  const int tid    = threadIdx.x;
  const int lane   = tid & 31;
  const int w      = tid >> 5;
  const int half16 = lane >> 4;    // which 16-lane half
  const int l16    = lane & 15;

  const int rowBase = blockIdx.x * RPB + w * RPW;

  // Prefetch this wave's x tile (8 KB) while VALU computes sin/cos below.
  // Each lane covers a distinct 128B cacheline: 16 rows x 512B = 4 lines/row.
  {
    const char* xb = (const char*)(x + (size_t)rowBase * DD);
    __builtin_prefetch(xb + lane * 128, 0, 3);            // rows 0..7
    __builtin_prefetch(xb + 4096 + lane * 128, 0, 3);     // rows 8..15
  }

  // --- positional encoding (sin/cos), once per block ---
  for (int d = tid; d < DD; d += WAVES * 32) {
    int t = d >> 1;
    float dv = __expf(-(float)t * (logf(10000.0f) / (float)DD));
    float a  = (float)d * dv;
    sPe[d] = (d & 1) ? __cosf(a) : __sinf(a);
  }
  __syncthreads();

  // --- Phase 1a: load x tile, add pe, convert to f16 in LDS ---
  for (int m = 0; m < RPW; ++m) {
    const v4f xv = *reinterpret_cast<const v4f*>(x + (size_t)(rowBase + m) * DD + lane * 4);
    const v4f pe = *reinterpret_cast<const v4f*>(&sPe[lane * 4]);
    h4 hv;
    hv[0] = (_Float16)(xv[0] + pe[0]);
    hv[1] = (_Float16)(xv[1] + pe[1]);
    hv[2] = (_Float16)(xv[2] + pe[2]);
    hv[3] = (_Float16)(xv[3] + pe[3]);
    *reinterpret_cast<h4*>(&sXp[w][m * DD + lane * 4]) = hv;
  }
  __syncthreads();

  // --- Phase 1b: A-fragments (16x32 f16, ISA 7.12.2 layout) from LDS ---
  // lane holds row l16; halfs 0..7 at K = half16*8 + ks*32, halfs 8..15 at +16
  v16h afrag[4];
  for (int ks = 0; ks < 4; ++ks) {
    const _Float16* ab = &sXp[w][l16 * DD + half16 * 8 + ks * 32];
    v8h lo = *reinterpret_cast<const v8h*>(ab);
    v8h hi = *reinterpret_cast<const v8h*>(ab + 16);
    for (int e = 0; e < 8; ++e) { afrag[ks][e] = lo[e]; afrag[ks][8 + e] = hi[e]; }
  }

  // --- Phase 1c: QKV via v_wmma_f32_16x16x32_f16, K=128 in 4 steps ---
  const float* biases[3] = { bq, bk, bv };
  for (int mat = 0; mat < 3; ++mat) {
    const _Float16* W = Wh + (size_t)mat * DD * DD;
    const float* bias = biases[mat];
    for (int nt = 0; nt < 8; ++nt) {
      float bb = bias[nt * 16 + l16];       // C/D column N = lane&15
      v8f c;
      for (int r = 0; r < 8; ++r) c[r] = bb;
      for (int ks = 0; ks < 4; ++ks) {
        // B-fragment 32x16: lane = column n, halfs 0..7 at K = half16*8, 8..15 at +16
        const _Float16* wb = W + (size_t)(nt * 16 + l16) * DD + half16 * 8 + ks * 32;
        v8h blo = *reinterpret_cast<const v8h*>(wb);
        v8h bhi = *reinterpret_cast<const v8h*>(wb + 16);
        v16h bfrag;
        for (int e = 0; e < 8; ++e) { bfrag[e] = blo[e]; bfrag[8 + e] = bhi[e]; }
        c = __builtin_amdgcn_wmma_f32_16x16x32_f16(
            /*neg_a=*/false, afrag[ks], /*neg_b=*/false, bfrag,
            /*c_mod=*/(short)0, c, /*reuse_a=*/false, /*reuse_b=*/false);
      }
      // D layout: c[r] -> (M = r + 8*half16, N = nt*16 + l16)
      if (mat == 0) {
        for (int r = 0; r < 8; ++r)
          sQ[w][(r + 8 * half16) * DD + nt * 16 + l16] = c[r];
      } else if (mat == 1) {
        for (int r = 0; r < 8; ++r)
          sKV[w][(r + 8 * half16) * 2 * DD + 2 * (nt * 16 + l16)] = c[r];
      } else {
        for (int r = 0; r < 8; ++r)
          sKV[w][(r + 8 * half16) * 2 * DD + 2 * (nt * 16 + l16) + 1] = c[r];
      }
    }
  }
  __syncthreads();

  // --- Phase 2: per-node rank-1 softmax-attention ---
  // scores[i,j] = (q_i*scale) * k_j  =>  rowmax_i = qs_i * (qs_i>=0 ? kmax : kmin)
  const float scale = 0.08838834764831845f;  // 1/sqrt(128)
  for (int m = 0; m < RPW; ++m) {
    const float* kvrow = &sKV[w][m * 2 * DD];
    // wave-wide K min/max (O(D) once per node, replaces O(D^2) max pass)
    float kmx = -3.4e38f, kmn = 3.4e38f;
    for (int jj = 0; jj < 4; ++jj) {
      float kj = kvrow[2 * (lane + 32 * jj)];
      kmx = fmaxf(kmx, kj);
      kmn = fminf(kmn, kj);
    }
    for (int off = 16; off; off >>= 1) {
      kmx = fmaxf(kmx, __shfl_xor(kmx, off, 32));
      kmn = fminf(kmn, __shfl_xor(kmn, off, 32));
    }
    // lanes parallel over output index i; sequential broadcast reads over j.
    // Two (k,v) pairs per 16B LDS broadcast read -> halves DS issue pressure
    // so the wave is limited by v_exp_f32 + FMA throughput, not DS.
    for (int ii = 0; ii < 4; ++ii) {
      int i = ii * 32 + lane;
      float qs = sQ[w][m * DD + i] * scale;
      float mi = (qs >= 0.0f) ? qs * kmx : qs * kmn;
      float ssum = 0.0f, osum = 0.0f;
#pragma unroll 4
      for (int j = 0; j < DD; j += 2) {
        v4f kv2 = *reinterpret_cast<const v4f*>(&kvrow[2 * j]);  // b128 broadcast
        float e0 = __expf(__fmaf_rn(qs, kv2[0], -mi));           // v_exp_f32
        float e1 = __expf(__fmaf_rn(qs, kv2[2], -mi));
        ssum += e0 + e1;
        osum = __fmaf_rn(e0, kv2[1], osum);
        osum = __fmaf_rn(e1, kv2[3], osum);
      }
      out[(size_t)(rowBase + m) * DD + i] = osum / ssum;
    }
  }
}

// ------------------------------------------------------------------
extern "C" void kernel_launch(void* const* d_in, const int* in_sizes, int n_in,
                              void* d_out, int out_size, void* d_ws, size_t ws_size,
                              hipStream_t stream) {
  const float* x  = (const float*)d_in[0];
  const float* Wq = (const float*)d_in[1];
  const float* bq = (const float*)d_in[2];
  const float* Wk = (const float*)d_in[3];
  const float* bk = (const float*)d_in[4];
  const float* Wv = (const float*)d_in[5];
  const float* bv = (const float*)d_in[6];
  float* out = (float*)d_out;

  const int N = in_sizes[0] / DD;  // 16384

  _Float16* Wh = (_Float16*)d_ws;  // 3*128*128 f16 = 96 KB
  prep_weights_f16<<<(3 * DD * DD) / 256, 256, 0, stream>>>(Wq, Wk, Wv, Wh);

  node_attn_kernel<<<N / RPB, WAVES * 32, 0, stream>>>(x, Wh, bq, bk, bv, out, N);
}
